// DbrxBlock_65816078844561
// MI455X (gfx1250) — matrix operand order repaired and checked by reading, not compiled, not run on hardware
//
#include <hip/hip_runtime.h>
#include <cstdint>
#include <cstddef>

// ---------------------------------------------------------------------------
// DBRX-style block for MI455X (gfx1250, wave32).  All GEMMs + attention run on
// v_wmma_f32_16x16x32_bf16 (bf16 in / fp32 accumulate): the block is WMMA
// compute bound (~850 GFLOP, dominated by the dense MoE exactly as in the
// reference), while unique memory traffic (~0.5 GB) is only ~22us at 23.3TB/s.
// The MoE dual-GEMM stages its fp32 expert weights through the Tensor Data
// Mover (tensor_load_to_lds + s_wait_tensorcnt), double-buffered in LDS.
// ---------------------------------------------------------------------------

typedef __attribute__((ext_vector_type(8)))  float  v8f;
typedef __attribute__((ext_vector_type(16))) __bf16 v16bf;
typedef __attribute__((ext_vector_type(8)))  __bf16 bf16x8;

#define DEV __device__ __forceinline__

// Problem dims
#define TOK    4096     // B*S
#define DMODEL 1024
#define NQKV   1536     // D + 2*KV*HD
#define NEXP   8
#define FF     4096

DEV v8f wmma_bf16(v16bf a, v16bf b, v8f c) {
  return __builtin_amdgcn_wmma_f32_16x16x32_bf16(false, a, false, b,
                                                 (short)0, c, false, false);
}

// A fragment (16x32 bf16, MxK): lane row m = lane&15;
// element j -> k = kbase + (j>>3)*16 + (lane>>4)*8 + (j&7)  => two 16B runs.
DEV v16bf frag_a(const __bf16* tile, int stride, int lane, int kbase) {
  const __bf16* p = tile + (size_t)(lane & 15) * stride + kbase + ((lane >> 4) << 3);
  bf16x8 lo = *(const bf16x8*)p;
  bf16x8 hi = *(const bf16x8*)(p + 16);
  v16bf r;
#pragma unroll
  for (int i = 0; i < 8; ++i) { r[i] = lo[i]; r[i + 8] = hi[i]; }
  return r;
}

// B fragment (32x16 bf16, KxN), LDS stored as [n][k]: lane col n = lane&15;
// element j -> k = kbase + (j>>3)*16 + (lane>>4)*8 + (j&7)  => two 16B runs.
DEV v16bf frag_b(const __bf16* tile, int stride, int lane, int kbase) {
  const __bf16* p = tile + (size_t)(lane & 15) * stride + kbase + ((lane >> 4) << 3);
  bf16x8 lo = *(const bf16x8*)p;
  bf16x8 hi = *(const bf16x8*)(p + 16);
  v16bf r;
#pragma unroll
  for (int i = 0; i < 8; ++i) { r[i] = lo[i]; r[i + 8] = hi[i]; }
  return r;
}

// ------------------- TDM: async tile load (64 rows x 32 cols fp32) ---------
// Builds a 2-group Tensor DMA Descriptor (CDNA5 ISA):
//  group0: count=1 | lds_addr | global_addr(tile start) | type=2
//  group1: data_size=4B, tensor 1024x4096, tile_dim0=32, tile_dim1=64,
//          tensor_dim0_stride=1024 (row-major [rows][1024] fp32 matrix).
// Issued once per wave (scalar op, EXEC-independent); tracked by TENSORcnt.
DEV void tdm_load_tile(const float* gsrc, unsigned lds_byte_off) {
  unsigned long long ga = (unsigned long long)gsrc;
  unsigned glo = (unsigned)ga;
  unsigned ghi = ((unsigned)(ga >> 32) & 0x01ffffffu) | 0x80000000u; // type=2
  asm volatile(
      "s_mov_b32 s36, 1\n\t"            // g0: count=1 (valid descriptor)
      "s_mov_b32 s37, %0\n\t"           // g0: lds_addr (bytes)
      "s_mov_b32 s38, %1\n\t"           // g0: global_addr[31:0]
      "s_mov_b32 s39, %2\n\t"           // g0: global_addr[56:32] | type<<30
      "s_mov_b32 s40, 0x00020000\n\t"   // g1: data_size=4B, mask=0
      "s_mov_b32 s41, 0x04000000\n\t"   // g1: tensor_dim0=1024 (bits 63:48)
      "s_mov_b32 s42, 0x10000000\n\t"   // g1: tensor_dim1=4096 (bits 95:80)
      "s_mov_b32 s43, 0x00200000\n\t"   // g1: tile_dim0=32 (bits 127:112)
      "s_mov_b32 s44, 64\n\t"           // g1: tile_dim1=64 (bits 143:128)
      "s_mov_b32 s45, 1024\n\t"         // g1: tensor_dim0_stride=1024
      "s_mov_b32 s46, 0\n\t"
      "s_mov_b32 s47, 0\n\t"
      "tensor_load_to_lds s[36:39], s[40:47]\n\t"
      :
      : "s"(lds_byte_off), "s"(glo), "s"(ghi)
      : "s36", "s37", "s38", "s39", "s40", "s41", "s42", "s43",
        "s44", "s45", "s46", "s47", "memory");
}

// ------------------------------ LayerNorm 1 --------------------------------
__global__ __launch_bounds__(256) void ln1_kernel(const float* __restrict__ x,
                                                  const float* __restrict__ scale,
                                                  __bf16* __restrict__ out) {
  int t = blockIdx.x, tid = threadIdx.x;
  const float* xp = x + (size_t)t * DMODEL;
  float v[4], s = 0.f, sq = 0.f;
#pragma unroll
  for (int i = 0; i < 4; ++i) { v[i] = xp[tid + i * 256]; s += v[i]; sq += v[i] * v[i]; }
#pragma unroll
  for (int o = 16; o; o >>= 1) { s += __shfl_xor(s, o, 32); sq += __shfl_xor(sq, o, 32); }
  __shared__ float red[18];
  int w = tid >> 5, lane = tid & 31;
  if (lane == 0) { red[w] = s; red[8 + w] = sq; }
  __syncthreads();
  if (tid == 0) {
    float S = 0.f, SQ = 0.f;
    for (int i = 0; i < 8; ++i) { S += red[i]; SQ += red[8 + i]; }
    red[16] = S; red[17] = SQ;
  }
  __syncthreads();
  float mu = red[16] * (1.f / DMODEL);
  float var = red[17] * (1.f / DMODEL) - mu * mu;
  float inv = rsqrtf(var + 1e-5f);
#pragma unroll
  for (int i = 0; i < 4; ++i) {
    int d = tid + i * 256;
    out[(size_t)t * DMODEL + d] = (__bf16)((v[i] - mu) * inv * scale[d]);
  }
}

// --------------------- LayerNorm 2 + router (top-2 gates) ------------------
__global__ __launch_bounds__(256) void ln2_router_kernel(
    const float* __restrict__ x, const float* __restrict__ scale,
    const float* __restrict__ rw, __bf16* __restrict__ h2,
    float* __restrict__ gates) {
  int t = blockIdx.x, tid = threadIdx.x;
  const float* xp = x + (size_t)t * DMODEL;
  float v[4], s = 0.f, sq = 0.f;
#pragma unroll
  for (int i = 0; i < 4; ++i) { v[i] = xp[tid + i * 256]; s += v[i]; sq += v[i] * v[i]; }
#pragma unroll
  for (int o = 16; o; o >>= 1) { s += __shfl_xor(s, o, 32); sq += __shfl_xor(sq, o, 32); }
  __shared__ float red[18];
  __shared__ float rbuf[256];
  __shared__ float logits[NEXP];
  int w = tid >> 5, lane = tid & 31;
  if (lane == 0) { red[w] = s; red[8 + w] = sq; }
  __syncthreads();
  if (tid == 0) {
    float S = 0.f, SQ = 0.f;
    for (int i = 0; i < 8; ++i) { S += red[i]; SQ += red[8 + i]; }
    red[16] = S; red[17] = SQ;
  }
  __syncthreads();
  float mu = red[16] * (1.f / DMODEL);
  float var = red[17] * (1.f / DMODEL) - mu * mu;
  float inv = rsqrtf(var + 1e-5f);
  float nv[4], acc[NEXP];
#pragma unroll
  for (int e = 0; e < NEXP; ++e) acc[e] = 0.f;
#pragma unroll
  for (int i = 0; i < 4; ++i) {
    int d = tid + i * 256;
    nv[i] = (v[i] - mu) * inv * scale[d];
    h2[(size_t)t * DMODEL + d] = (__bf16)nv[i];
    const float* r = rw + (size_t)d * NEXP;
#pragma unroll
    for (int e = 0; e < NEXP; ++e) acc[e] += nv[i] * r[e];
  }
  for (int e = 0; e < NEXP; ++e) {
    rbuf[tid] = acc[e];
    __syncthreads();
    for (int st = 128; st; st >>= 1) {
      if (tid < st) rbuf[tid] += rbuf[tid + st];
      __syncthreads();
    }
    if (tid == 0) logits[e] = rbuf[0];
    __syncthreads();
  }
  if (tid == 0) {
    float mx = logits[0];
    for (int e = 1; e < NEXP; ++e) mx = fmaxf(mx, logits[e]);
    float p[NEXP], sum = 0.f;
    for (int e = 0; e < NEXP; ++e) { p[e] = __expf(logits[e] - mx); sum += p[e]; }
    float isum = 1.f / sum;
    for (int e = 0; e < NEXP; ++e) p[e] *= isum;
    int e0 = 0; float b0 = p[0];
    for (int e = 1; e < NEXP; ++e) if (p[e] > b0) { b0 = p[e]; e0 = e; }
    int e1 = -1; float b1 = -1.f;
    for (int e = 0; e < NEXP; ++e) if (e != e0 && p[e] > b1) { b1 = p[e]; e1 = e; }
    float inv2 = 1.f / (b0 + b1);
    float g[NEXP];
    for (int e = 0; e < NEXP; ++e) g[e] = 0.f;
    g[e0] = b0 * inv2; g[e1] = b1 * inv2;
    for (int e = 0; e < NEXP; ++e) gates[(size_t)t * NEXP + e] = g[e];
  }
}

// --------------------------- RoPE + GQA split ------------------------------
__global__ __launch_bounds__(256) void rope_split_kernel(
    const float* __restrict__ qkv, const float* __restrict__ sin_t,
    const float* __restrict__ cos_t, const int* __restrict__ pos_ids,
    __bf16* __restrict__ q, __bf16* __restrict__ k, __bf16* __restrict__ v) {
  int t = blockIdx.x, tid = threadIdx.x;
  int b = t >> 10, s = t & 1023;
  int pos = pos_ids[t];
  const float* row = qkv + (size_t)t * NQKV;
#pragma unroll
  for (int i = 0; i < 4; ++i) {
    int c = tid + i * 256;
    int h = c >> 6, d = c & 63;
    float sn = sin_t[pos * 64 + d], cs = cos_t[pos * 64 + d];
    float val = row[c];
    float other = (d < 32) ? -row[c + 32] : row[c - 32];
    q[((size_t)(b * 16 + h) * 1024 + s) * 64 + d] = (__bf16)(val * cs + other * sn);
  }
  {
    int c = 1024 + tid, kv = tid >> 6, d = tid & 63;
    float sn = sin_t[pos * 64 + d], cs = cos_t[pos * 64 + d];
    float val = row[c];
    float other = (d < 32) ? -row[c + 32] : row[c - 32];
    k[((size_t)(b * 4 + kv) * 1024 + s) * 64 + d] = (__bf16)(val * cs + other * sn);
  }
  {
    int c = 1280 + tid, kv = tid >> 6, d = tid & 63;
    v[((size_t)(b * 4 + kv) * 1024 + s) * 64 + d] = (__bf16)row[c];
  }
}

// ----------------------- Generic WMMA GEMM (B is fp32, [K][N]) -------------
// BM=128, BN=128, BK=32; 8 waves (4x2), each wave computes a 32x64 tile
// (2x4 WMMA frags): 8 WMMAs per 6 fragment loads per k-step.
enum { EPI_F32 = 0, EPI_RES = 1, EPI_GATED = 2 };

template <int EPI>
__global__ __launch_bounds__(256) void gemm_kernel(
    const __bf16* __restrict__ A, const float* __restrict__ Bg,
    int M, int N, int K, int ldb,
    float* __restrict__ C0, float* __restrict__ C1,
    const float* __restrict__ addsrc, const float* __restrict__ gates) {
  __shared__ __align__(16) __bf16 As[128][40];
  __shared__ __align__(16) __bf16 Bs[128][40];
  int tid = threadIdx.x, lane = tid & 31, w = tid >> 5;
  int nb = blockIdx.x, mb = blockIdx.y;
  int wm = (w & 3) * 32, wn = (w >> 2) * 64;
  v8f acc[2][4] = {};
  const int arow = tid >> 1, acol = (tid & 1) * 16;
  for (int kb = 0; kb < K; kb += 32) {
    __syncthreads();
    {
      const __bf16* src = A + (size_t)(mb * 128 + arow) * K + kb + acol;
      *(bf16x8*)&As[arow][acol] = *(const bf16x8*)src;
      *(bf16x8*)&As[arow][acol + 8] = *(const bf16x8*)(src + 8);
      if (kb + 32 < K) __builtin_prefetch(src + 32, 0, 1);  // global_prefetch_b8
    }
    {
      // B stored row-major [K][N]; stage transposed+bf16 into Bs[n][k].
      int k = tid >> 3, n0 = (tid & 7) * 16;
      const float* src = Bg + (size_t)(kb + k) * ldb + nb * 128 + n0;
#pragma unroll
      for (int j = 0; j < 16; ++j) Bs[n0 + j][k] = (__bf16)src[j];
    }
    __syncthreads();
    v16bf a0 = frag_a(&As[wm][0], 40, lane, 0);
    v16bf a1 = frag_a(&As[wm + 16][0], 40, lane, 0);
    v16bf b[4];
#pragma unroll
    for (int j = 0; j < 4; ++j) b[j] = frag_b(&Bs[wn + j * 16][0], 40, lane, 0);
#pragma unroll
    for (int sn = 0; sn < 4; ++sn) {
      acc[0][sn] = wmma_bf16(a0, b[sn], acc[0][sn]);
      acc[1][sn] = wmma_bf16(a1, b[sn], acc[1][sn]);
    }
  }
  int gm = mb * 128 + wm, gn = nb * 128 + wn;
#pragma unroll
  for (int sm = 0; sm < 2; ++sm)
#pragma unroll
    for (int sn = 0; sn < 4; ++sn) {
      int n = gn + sn * 16 + (lane & 15);
#pragma unroll
      for (int i = 0; i < 8; ++i) {
        int m = gm + sm * 16 + i + ((lane >> 4) << 3);
        size_t idx = (size_t)m * N + n;
        float val = acc[sm][sn][i];
        if constexpr (EPI == EPI_F32) {
          C0[idx] = val;
        } else if constexpr (EPI == EPI_RES) {
          float r = val + addsrc[idx];
          C0[idx] = r;
          C1[idx] = r;
        } else {
          float g = gates[(size_t)m * NEXP];
          if (g != 0.f) C0[idx] += g * val;
        }
      }
    }
}

// ----------------- Fused dual GEMM: x1 = silu(h2 W1^T) * (h2 V1^T) ---------
// W1/V1 stored [F][D] row-major.  Expert weight tiles (64x32 fp32) are staged
// into LDS by the Tensor Data Mover, double-buffered: while tile k feeds the
// WMMAs, the TDM streams tile k+1 (wave 0 issues; s_wait_tensorcnt 0 drains
// before the publishing barrier).  Vector pipes only carry A + converts.
__global__ __launch_bounds__(256) void gemm_swiglu_kernel(
    const __bf16* __restrict__ A, const float* __restrict__ W1,
    const float* __restrict__ V1, __bf16* __restrict__ X1,
    int M, int N, int K, int ldb) {
  __shared__ __align__(16) __bf16 As[128][40];
  __shared__ __align__(16) __bf16 Bw[64][40];
  __shared__ __align__(16) __bf16 Bv[64][40];
  __shared__ __align__(16) float Sw[2][64 * 32];  // TDM landing (double buf)
  __shared__ __align__(16) float Sv[2][64 * 32];
  int tid = threadIdx.x, lane = tid & 31, w = tid >> 5;
  int nb = blockIdx.x, mb = blockIdx.y;
  int wm = (w & 3) * 32, wn = (w >> 2) * 32;
  v8f cw[2][2] = {}, cv[2][2] = {};
  const int arow = tid >> 1, acol = (tid & 1) * 16;
  const float* wbase = W1 + (size_t)(nb * 64) * ldb;
  const float* vbase = V1 + (size_t)(nb * 64) * ldb;
  if (w == 0) {
    tdm_load_tile(wbase, (unsigned)(size_t)&Sw[0][0]);
    tdm_load_tile(vbase, (unsigned)(size_t)&Sv[0][0]);
  }
  for (int kb = 0; kb < K; kb += 32) {
    int st = (kb >> 5) & 1;
    if (w == 0) __builtin_amdgcn_s_wait_tensorcnt(0);  // stage st landed
    __syncthreads();
    {
      const __bf16* src = A + (size_t)(mb * 128 + arow) * K + kb + acol;
      *(bf16x8*)&As[arow][acol] = *(const bf16x8*)src;
      *(bf16x8*)&As[arow][acol + 8] = *(const bf16x8*)(src + 8);
    }
    {
      // Convert TDM-staged fp32 weight tiles -> bf16 [n][k] operand tiles.
      int n = tid >> 2, c = (tid & 3) * 8;
      const float* s1 = &Sw[st][n * 32 + c];
      const float* s2 = &Sv[st][n * 32 + c];
      bf16x8 d1, d2;
#pragma unroll
      for (int j = 0; j < 8; ++j) { d1[j] = (__bf16)s1[j]; d2[j] = (__bf16)s2[j]; }
      *(bf16x8*)&Bw[n][c] = d1;
      *(bf16x8*)&Bv[n][c] = d2;
    }
    if (w == 0 && kb + 32 < K) {  // async-fetch next k-tile into other stage
      tdm_load_tile(wbase + kb + 32, (unsigned)(size_t)&Sw[st ^ 1][0]);
      tdm_load_tile(vbase + kb + 32, (unsigned)(size_t)&Sv[st ^ 1][0]);
    }
    __syncthreads();
    v16bf a0 = frag_a(&As[wm][0], 40, lane, 0);
    v16bf a1 = frag_a(&As[wm + 16][0], 40, lane, 0);
    v16bf bw0 = frag_b(&Bw[wn][0], 40, lane, 0);
    v16bf bw1 = frag_b(&Bw[wn + 16][0], 40, lane, 0);
    v16bf bv0 = frag_b(&Bv[wn][0], 40, lane, 0);
    v16bf bv1 = frag_b(&Bv[wn + 16][0], 40, lane, 0);
    cw[0][0] = wmma_bf16(a0, bw0, cw[0][0]);
    cw[0][1] = wmma_bf16(a0, bw1, cw[0][1]);
    cw[1][0] = wmma_bf16(a1, bw0, cw[1][0]);
    cw[1][1] = wmma_bf16(a1, bw1, cw[1][1]);
    cv[0][0] = wmma_bf16(a0, bv0, cv[0][0]);
    cv[0][1] = wmma_bf16(a0, bv1, cv[0][1]);
    cv[1][0] = wmma_bf16(a1, bv0, cv[1][0]);
    cv[1][1] = wmma_bf16(a1, bv1, cv[1][1]);
  }
  int gm = mb * 128 + wm, gn = nb * 64 + wn;
#pragma unroll
  for (int sm = 0; sm < 2; ++sm)
#pragma unroll
    for (int sn = 0; sn < 2; ++sn) {
      int n = gn + sn * 16 + (lane & 15);
#pragma unroll
      for (int i = 0; i < 8; ++i) {
        int m = gm + sm * 16 + i + ((lane >> 4) << 3);
        float x = cw[sm][sn][i];
        float val = x * (1.f / (1.f + __expf(-x))) * cv[sm][sn][i];
        X1[(size_t)m * N + n] = (__bf16)val;
      }
    }
}

// ------------------------- Flash attention (causal GQA) --------------------
// Block: 128 q-rows x full head; 8 waves, each wave owns 16 q-rows.
// QK^T and P*V both via WMMA; online softmax via xor-shuffle row reductions
// matched to the documented 16x16 f32 C/D lane layout.
__global__ __launch_bounds__(256) void attn_kernel(
    const __bf16* __restrict__ Q, const __bf16* __restrict__ Kg,
    const __bf16* __restrict__ Vg, __bf16* __restrict__ O) {
  __shared__ __align__(16) __bf16 Qs[128][72];
  __shared__ __align__(16) __bf16 Ks[32][72];
  __shared__ __align__(16) __bf16 Vs[64][40];   // transposed: Vs[d][kv_row]
  __shared__ __align__(16) __bf16 Ps[8][16][40];
  int tid = threadIdx.x, lane = tid & 31, w = tid >> 5;
  int qtile = blockIdx.x, bh = blockIdx.y;
  int b = bh >> 4, h = bh & 15, kvh = h >> 2;
  const __bf16* Qp = Q + (size_t)bh * 1024 * 64;
  const __bf16* Kp = Kg + (size_t)(b * 4 + kvh) * 1024 * 64;
  const __bf16* Vp = Vg + (size_t)(b * 4 + kvh) * 1024 * 64;
  int qbase = qtile * 128;
  for (int i = tid; i < 1024; i += 256) {
    int r = i >> 3, c = (i & 7) * 8;
    *(bf16x8*)&Qs[r][c] = *(const bf16x8*)&Qp[(size_t)(qbase + r) * 64 + c];
  }
  float rm[8], rs[8];
  v8f o[4] = {};
#pragma unroll
  for (int i = 0; i < 8; ++i) { rm[i] = -1e30f; rs[i] = 0.f; }
  int half = lane >> 4;
  int qrow0 = qbase + w * 16 + half * 8;
  for (int kbb = 0; kbb < qbase + 128; kbb += 32) {
    __syncthreads();
    {
      int r = tid >> 3, c = (tid & 7) * 8;
      *(bf16x8*)&Ks[r][c] = *(const bf16x8*)&Kp[(size_t)(kbb + r) * 64 + c];
    }
    {
      int d = tid >> 2, r0 = (tid & 3) * 8;
#pragma unroll
      for (int j = 0; j < 8; ++j)
        Vs[d][r0 + j] = Vp[(size_t)(kbb + r0 + j) * 64 + d];
    }
    __syncthreads();
    v8f s0 = {}, s1 = {};
#pragma unroll
    for (int kk = 0; kk < 64; kk += 32) {
      v16bf aq = frag_a(&Qs[w * 16][0], 72, lane, kk);
      s0 = wmma_bf16(aq, frag_b(&Ks[0][0], 72, lane, kk), s0);
      s1 = wmma_bf16(aq, frag_b(&Ks[16][0], 72, lane, kk), s1);
    }
    int kc0 = kbb + (lane & 15);
#pragma unroll
    for (int i = 0; i < 8; ++i) {
      int qr = qrow0 + i;
      float v0 = s0[i] * 0.125f; if (kc0 > qr) v0 = -1e9f;
      float v1 = s1[i] * 0.125f; if (kc0 + 16 > qr) v1 = -1e9f;
      float mx = fmaxf(v0, v1);
#pragma unroll
      for (int om = 8; om; om >>= 1) mx = fmaxf(mx, __shfl_xor(mx, om, 32));
      float nm = fmaxf(rm[i], mx);
      float alpha = __expf(rm[i] - nm);
      float p0 = __expf(v0 - nm), p1 = __expf(v1 - nm);
      float ps = p0 + p1;
#pragma unroll
      for (int om = 8; om; om >>= 1) ps += __shfl_xor(ps, om, 32);
      rs[i] = rs[i] * alpha + ps;
      rm[i] = nm;
#pragma unroll
      for (int dt = 0; dt < 4; ++dt) o[dt][i] = o[dt][i] * alpha;
      int prow = half * 8 + i;
      Ps[w][prow][lane & 15] = (__bf16)p0;
      Ps[w][prow][(lane & 15) + 16] = (__bf16)p1;
    }
    // Wave-private LDS round-trip (D-layout -> A-layout); same-wave RAW is
    // covered by compiler-inserted s_wait_dscnt.
    v16bf ap = frag_a(&Ps[w][0][0], 40, lane, 0);
#pragma unroll
    for (int dt = 0; dt < 4; ++dt)
      o[dt] = wmma_bf16(ap, frag_b(&Vs[dt * 16][0], 40, lane, 0), o[dt]);
  }
#pragma unroll
  for (int i = 0; i < 8; ++i) rs[i] = 1.f / rs[i];
#pragma unroll
  for (int dt = 0; dt < 4; ++dt) {
    int cn = h * 64 + dt * 16 + (lane & 15);
#pragma unroll
    for (int i = 0; i < 8; ++i) {
      int m = qrow0 + i;
      O[(size_t)(b * 1024 + m) * 1024 + cn] = (__bf16)(o[dt][i] * rs[i]);
    }
  }
}

// ---------------------------------------------------------------------------
extern "C" void kernel_launch(void* const* d_in, const int* in_sizes, int n_in,
                              void* d_out, int out_size, void* d_ws, size_t ws_size,
                              hipStream_t stream) {
  const float* hidden = (const float*)d_in[0];
  const int*   pos    = (const int*)d_in[1];
  const float* ln1s   = (const float*)d_in[2];
  const float* ln2s   = (const float*)d_in[3];
  const float* wqkv   = (const float*)d_in[4];
  const float* wo     = (const float*)d_in[5];
  const float* rw     = (const float*)d_in[6];
  const float* w1     = (const float*)d_in[7];
  const float* v1     = (const float*)d_in[8];
  const float* w2     = (const float*)d_in[9];
  const float* rsin   = (const float*)d_in[10];
  const float* rcos   = (const float*)d_in[11];
  float* out = (float*)d_out;

  // Workspace layout (~114 MB total)
  char* ws = (char*)d_ws;
  size_t off = 0;
  auto take = [&](size_t bytes) {
    char* p = ws + off;
    off += (bytes + 255) & ~(size_t)255;
    return p;
  };
  __bf16* h1   = (__bf16*)take((size_t)TOK * DMODEL * 2);
  float*  qkv  = (float*) take((size_t)TOK * NQKV * 4);
  __bf16* qb   = (__bf16*)take((size_t)TOK * DMODEL * 2);
  __bf16* kbuf = (__bf16*)take((size_t)TOK * 256 * 2);
  __bf16* vbuf = (__bf16*)take((size_t)TOK * 256 * 2);
  __bf16* ao   = (__bf16*)take((size_t)TOK * DMODEL * 2);
  float*  resid= (float*) take((size_t)TOK * DMODEL * 4);
  __bf16* h2   = (__bf16*)take((size_t)TOK * DMODEL * 2);
  float*  gates= (float*) take((size_t)TOK * NEXP * 4);
  __bf16* x1   = (__bf16*)take((size_t)TOK * FF * 2);
  (void)ws_size; (void)in_sizes; (void)n_in; (void)out_size;

  // 1) LN1 -> bf16
  ln1_kernel<<<TOK, 256, 0, stream>>>(hidden, ln1s, h1);
  // 2) QKV GEMM (fp32 out, pre-RoPE)
  gemm_kernel<EPI_F32><<<dim3(NQKV / 128, TOK / 128), 256, 0, stream>>>(
      h1, wqkv, TOK, NQKV, DMODEL, NQKV, qkv, nullptr, nullptr, nullptr);
  // 3) RoPE + GQA split to [B,H,S,64] / [B,KV,S,64] bf16
  rope_split_kernel<<<TOK, 256, 0, stream>>>(qkv, rsin, rcos, pos, qb, kbuf, vbuf);
  // 4) Flash attention
  attn_kernel<<<dim3(8, 64), 256, 0, stream>>>(qb, kbuf, vbuf, ao);
  // 5) out-proj + residual -> resid (ws) and d_out (MoE accum base)
  gemm_kernel<EPI_RES><<<dim3(DMODEL / 128, TOK / 128), 256, 0, stream>>>(
      ao, wo, TOK, DMODEL, DMODEL, DMODEL, resid, out, hidden, nullptr);
  // 6) LN2 + router top-2 gates
  ln2_router_kernel<<<TOK, 256, 0, stream>>>(resid, ln2s, rw, h2, gates);
  // 7) MoE: per expert (sequential -> deterministic FP accumulation order)
  for (int e = 0; e < NEXP; ++e) {
    const size_t eo = (size_t)e * FF * DMODEL;
    gemm_swiglu_kernel<<<dim3(FF / 64, TOK / 128), 256, 0, stream>>>(
        h2, w1 + eo, v1 + eo, x1, TOK, FF, DMODEL, DMODEL);
    gemm_kernel<EPI_GATED><<<dim3(DMODEL / 128, TOK / 128), 256, 0, stream>>>(
        x1, w2 + eo, TOK, DMODEL, FF, DMODEL, out, nullptr, nullptr, gates + e);
  }
}
